// DeepFMImp_59081570123972
// MI455X (gfx1250) — matrix operand order repaired and checked by reading, not compiled, not run on hardware
//
#include <hip/hip_runtime.h>
#include <math.h>

typedef __attribute__((ext_vector_type(16))) __bf16          v16bf;
typedef __attribute__((ext_vector_type(8)))  float           v8f;

#define EDIM    16
#define F_NUM   13
#define F_CAT   26
#define NFIELDS 41              // 2 + 13 + 26
#define KREAL   (NFIELDS * EDIM)   // 656
#define KPAD    672             // pad to 21 * 32
#define KSTEPS  21
#define BATCH   65536

// OFFSETS = cumsum of CAT_VOCAB (exclusive)
__constant__ int d_off[F_CAT] = {
    0, 1000, 1500, 3500, 3600, 3650, 13650, 18650, 18950, 19150,
    20650, 21450, 21850, 22450, 24950, 25070, 25130, 28130, 28830,
    29730, 30830, 32130, 32380, 32730, 33180, 33730};

__device__ __forceinline__ void load8(const float* __restrict__ p, float* v) {
    float4 a = ((const float4*)p)[0];
    float4 b = ((const float4*)p)[1];
    v[0] = a.x; v[1] = a.y; v[2] = a.z; v[3] = a.w;
    v[4] = b.x; v[5] = b.y; v[6] = b.z; v[7] = b.w;
}

// Load 8 consecutive embedding elements (elems eb..eb+7) of field f for sample s.
__device__ __forceinline__ void load_chunk(
    int f, int eb, int s,
    const float* __restrict__ urow, const float* __restrict__ irow,
    const float* __restrict__ numeric, const float* __restrict__ numW,
    const float* __restrict__ numB, const int* __restrict__ catf,
    const float* __restrict__ cat_emb, float* v)
{
    if (f == 0) {
        load8(urow + eb, v);
    } else if (f == 1) {
        load8(irow + eb, v);
    } else if (f < 2 + F_NUM) {
        int j = f - 2;
        float x = numeric[s * F_NUM + j];
        float w[8], c[8];
        load8(numW + j * EDIM + eb, w);
        load8(numB + j * EDIM + eb, c);
#pragma unroll
        for (int i = 0; i < 8; ++i) v[i] = fmaf(x, w[i], c[i]);
    } else if (f < NFIELDS) {
        int j  = f - (2 + F_NUM);
        int ci = catf[s * F_CAT + j] + d_off[j];
        load8(cat_emb + ci * EDIM + eb, v);
    } else {                                 // K padding field
#pragma unroll
        for (int i = 0; i < 8; ++i) v[i] = 0.f;
    }
}

// ---- Kernel 1: transpose+pad mlp_W0 (656x16 f32, row-major [K][N]) to bf16 [N=16][KPAD]
__global__ __launch_bounds__(256) void convert_w0_kernel(
    const float* __restrict__ W0, unsigned short* __restrict__ w0t)
{
    int idx = blockIdx.x * 256 + threadIdx.x;
    if (idx >= 16 * KPAD) return;
    int n = idx / KPAD;
    int k = idx % KPAD;
    float v = (k < KREAL) ? W0[k * 16 + n] : 0.f;
    __bf16 h = (__bf16)v;                  // native f32->bf16 convert
    w0t[n * KPAD + k] = __builtin_bit_cast(unsigned short, h);
}

// ---- Kernel 2: fully fused DeepFM forward.
// 256 threads = 8 waves; each wave owns a 16-sample tile (128 samples/block).
__global__ __launch_bounds__(256) void deepfm_fused_kernel(
    const int*   __restrict__ user_input, const int*   __restrict__ item_input,
    const float* __restrict__ numeric,    const int*   __restrict__ catf,
    const float* __restrict__ user_emb,   const float* __restrict__ item_emb,
    const float* __restrict__ numW,       const float* __restrict__ numB,
    const float* __restrict__ cat_emb,    const unsigned short* __restrict__ w0t,
    const float* __restrict__ b0,
    const float* __restrict__ W1, const float* __restrict__ b1,
    const float* __restrict__ W2, const float* __restrict__ b2,
    const float* __restrict__ W3, const float* __restrict__ b3,
    const float* __restrict__ fm_sW, const float* __restrict__ fm_sb,
    const float* __restrict__ fm_lW, const float* __restrict__ fm_lb,
    const float* __restrict__ fin_W, const float* __restrict__ fin_b,
    float* __restrict__ out)
{
    __shared__ float lds_h0[8][16][16];    // per-wave 16x16 layer-0 activations

    const int lane = threadIdx.x & 31;
    const int wave = threadIdx.x >> 5;
    const int half = lane >> 4;            // 0: K-offsets 0-7/16-23, 1: 8-15/24-31
    const int m    = lane & 15;            // A-matrix row = sample within tile
    const int sampleBase = (blockIdx.x * 8 + wave) * 16;
    const int s  = sampleBase + m;
    const int eb = half * 8;               // embedding-element base this lane covers

    const float* urow = user_emb + (long)user_input[s] * EDIM;
    const float* irow = item_emb + (long)item_input[s] * EDIM;

    // FM accumulators: per-e sums over fields (this lane's 8 e-positions) + sum of squares
    float sumv[8];
#pragma unroll
    for (int i = 0; i < 8; ++i) sumv[i] = 0.f;
    float sumsq = 0.f;

    v8f acc = {};                          // 16x16 f32 accumulator (layer 0)
    const int n = m;                       // B-matrix column this lane supplies
    const unsigned short* brow = w0t + n * KPAD + half * 8;

    for (int kb = 0; kb < KSTEPS; ++kb) {
        float v0[8], v1[8];
        load_chunk(2 * kb,     eb, s, urow, irow, numeric, numW, numB, catf, cat_emb, v0);
        load_chunk(2 * kb + 1, eb, s, urow, irow, numeric, numW, numB, catf, cat_emb, v1);

#pragma unroll
        for (int i = 0; i < 8; ++i) {
            sumv[i] += v0[i] + v1[i];
            sumsq = fmaf(v0[i], v0[i], sumsq);
            sumsq = fmaf(v1[i], v1[i], sumsq);
        }

        // Pack A fragment with native converts:
        // VGPR0-3 = this lane's first K-octet, VGPR4-7 = second.
        v16bf a;
#pragma unroll
        for (int i = 0; i < 8; ++i) {
            a[i]     = (__bf16)v0[i];
            a[8 + i] = (__bf16)v1[i];
        }
        v16bf b;
        const __bf16* bp = (const __bf16*)(brow + kb * 32);
#pragma unroll
        for (int i = 0; i < 8; ++i) {
            b[i]     = bp[i];
            b[8 + i] = bp[16 + i];
        }

        acc = __builtin_amdgcn_wmma_f32_16x16x32_bf16(
            /*neg_a=*/false, a, /*neg_b=*/false, b,
            /*c_mod=*/(short)0, acc, /*reuse_a=*/false, /*reuse_b=*/false);
    }

    // layer-0 bias + relu, transpose via LDS (D layout: lane n holds M=r+half*8 in acc[r])
    const float bn = b0[n];
#pragma unroll
    for (int r = 0; r < 8; ++r) {
        float h = acc[r] + bn;
        lds_h0[wave][r + half * 8][n] = h > 0.f ? h : 0.f;
    }

    // FM second-order: 0.5 * (sum_e sum_f^2 - sum_{f,e} x^2), halves combined via shfl
    float se = 0.f;
#pragma unroll
    for (int i = 0; i < 8; ++i) se = fmaf(sumv[i], sumv[i], se);
    se += __shfl_xor(se, 16);
    float sq = sumsq + __shfl_xor(sumsq, 16);
    float fm_embed = 0.5f * (se - sq);

    // FM first-order sparse: half 0 does numeric dot, half 1 does categorical gather-sum
    float sp = 0.f;
    if (half == 0) {
#pragma unroll
        for (int j = 0; j < F_NUM; ++j)
            sp = fmaf(numeric[s * F_NUM + j], fm_sW[j], sp);
    } else {
        for (int j = 0; j < F_CAT; ++j)
            sp += fm_sW[F_NUM + catf[s * F_CAT + j] + d_off[j]];
    }
    sp += __shfl_xor(sp, 16);
    sp += fm_sb[0];
    float y_fm = fm_embed * fm_lW[0] + sp * fm_lW[1] + fm_lb[0];

    __syncthreads();

    // Tiny MLP tail (16->32->64->1), one sample per lane on the low half-wave
    if (half == 0) {
        float h0v[16];
#pragma unroll
        for (int i = 0; i < 16; ++i) h0v[i] = lds_h0[wave][m][i];

        float h1[32];
#pragma unroll
        for (int o = 0; o < 32; ++o) {
            float a1 = b1[o];
#pragma unroll
            for (int i = 0; i < 16; ++i) a1 = fmaf(h0v[i], W1[i * 32 + o], a1);
            h1[o] = a1 > 0.f ? a1 : 0.f;
        }

        float yd = b3[0];
        for (int o = 0; o < 64; ++o) {
            float a2 = b2[o];
#pragma unroll
            for (int i = 0; i < 32; ++i) a2 = fmaf(h1[i], W2[i * 64 + o], a2);
            a2 = a2 > 0.f ? a2 : 0.f;
            yd = fmaf(a2, W3[o], yd);
        }
        yd = yd > 0.f ? yd : 0.f;

        float z = y_fm * fin_W[0] + yd * fin_W[1] + fin_b[0];
        out[s] = 1.f / (1.f + __expf(-z));
    }
}

extern "C" void kernel_launch(void* const* d_in, const int* in_sizes, int n_in,
                              void* d_out, int out_size, void* d_ws, size_t ws_size,
                              hipStream_t stream) {
    const int*   user_input = (const int*)  d_in[0];
    const int*   item_input = (const int*)  d_in[1];
    const float* numeric    = (const float*)d_in[2];
    const int*   catf       = (const int*)  d_in[3];
    const float* user_emb   = (const float*)d_in[4];
    const float* item_emb   = (const float*)d_in[5];
    const float* num_W      = (const float*)d_in[6];
    const float* num_b      = (const float*)d_in[7];
    const float* cat_emb    = (const float*)d_in[8];
    const float* W0         = (const float*)d_in[9];
    const float* b0         = (const float*)d_in[10];
    const float* W1         = (const float*)d_in[11];
    const float* b1         = (const float*)d_in[12];
    const float* W2         = (const float*)d_in[13];
    const float* b2         = (const float*)d_in[14];
    const float* W3         = (const float*)d_in[15];
    const float* b3         = (const float*)d_in[16];
    const float* fm_sW      = (const float*)d_in[17];
    const float* fm_sb      = (const float*)d_in[18];
    const float* fm_lW      = (const float*)d_in[19];
    const float* fm_lb      = (const float*)d_in[20];
    const float* fin_W      = (const float*)d_in[21];
    const float* fin_b      = (const float*)d_in[22];

    unsigned short* w0t = (unsigned short*)d_ws;   // 16*672 bf16 = 21504 B

    convert_w0_kernel<<<(16 * KPAD + 255) / 256, 256, 0, stream>>>(W0, w0t);

    deepfm_fused_kernel<<<BATCH / 128, 256, 0, stream>>>(
        user_input, item_input, numeric, catf,
        user_emb, item_emb, num_W, num_b, cat_emb,
        w0t, b0, W1, b1, W2, b2, W3, b3,
        fm_sW, fm_sb, fm_lW, fm_lb, fin_W, fin_b,
        (float*)d_out);
}
